// CenMoERewardModel_68358699483592
// MI455X (gfx1250) — matrix operand order repaired and checked by reading, not compiled, not run on hardware
//
#include <hip/hip_runtime.h>
#include <hip/hip_bf16.h>

// ---------------------------------------------------------------------------
// CDNA5 (gfx1250) WMMA implementation of CenMoERewardModel forward.
// wave32; v_wmma_f32_16x16x32_f16 for all GEMMs; TDM async tensor load for
// the per-token activation tile.
// ---------------------------------------------------------------------------

typedef __attribute__((ext_vector_type(16))) _Float16 v16h;
typedef __attribute__((ext_vector_type(8)))  float    v8f;
typedef __attribute__((ext_vector_type(4)))  unsigned v4u;
typedef __attribute__((ext_vector_type(8)))  int      v8i;
typedef __attribute__((ext_vector_type(4)))  int      v4i;

#define DEV static __device__ __forceinline__

// Model dims
#define BB   1024
#define NA   32
#define DD   128     // dz+da
#define DZ   96
#define DA   32
#define EE   8
#define FF   1024
#define HH   512
#define NRR  4
#define NH   4
#define HD   32
#define TOKD (NA*DD) // 4096

// --- TDM availability / arity guards ---------------------------------------
#if defined(__has_builtin)
#if __has_builtin(__builtin_amdgcn_tensor_load_to_lds) && __has_builtin(__builtin_amdgcn_s_wait_tensorcnt)
#define HAVE_TDM 1
#endif
#endif

#if defined(HAVE_TDM)
#if __has_include(<hip/amd_detail/amd_gfx1250_TDM.h>)
// therock / clang-23: 6-arg form
#define TDM_LOAD(g0, g1, g2, g3) \
  __builtin_amdgcn_tensor_load_to_lds((g0), (g1), (g2), (g3), (v8i){0,0,0,0,0,0,0,0}, 0)
#else
// ROCm 7.2 / clang-22: 5-arg form
#define TDM_LOAD(g0, g1, g2, g3) \
  __builtin_amdgcn_tensor_load_to_lds((g0), (g1), (g2), (g3), 0)
#endif
#endif

DEV v8f wmma16(v16h a, v16h b, v8f c) {
  return __builtin_amdgcn_wmma_f32_16x16x32_f16(
      /*neg_a=*/false, a, /*neg_b=*/false, b,
      /*c_mod=*/(short)0, c, /*reuse_a=*/false, /*reuse_b=*/false);
}

// --- WMMA operand loaders (contiguous k = 16*(lane/16) + s, wave32) -----------
// A tile: 16x32 f16 from row-major [*, ld]; row m = lane%16;
// lanes>=16 add +16 to k; VGPR v holds k=2v,+1.
DEV v16h load_A(const _Float16* p, int ld, int row0, int col0) {
  int lane = threadIdx.x & 31;
  const _Float16* rp = p + (row0 + (lane & 15)) * ld + col0 + ((lane >> 4) << 4);
  v16h a;
#pragma unroll
  for (int v = 0; v < 8; ++v) {
    int k = v << 1;
    a[2 * v]     = rp[k];
    a[2 * v + 1] = rp[k + 1];
  }
  return a;
}

// B tile: 32x16 (KxN) where the source matrix is row-major [N,K] (a weight
// for out = in @ W^T).  col n = lane%16; lanes>=16 add +16 to k; VGPR v: k=2v.
DEV v16h load_B_nk(const _Float16* p, int ld, int n0, int k0) {
  int lane = threadIdx.x & 31;
  const _Float16* rp = p + (n0 + (lane & 15)) * ld + k0 + ((lane >> 4) << 4);
  v16h b;
#pragma unroll
  for (int v = 0; v < 8; ++v) { b[2 * v] = rp[2 * v]; b[2 * v + 1] = rp[2 * v + 1]; }
  return b;
}

// Same but zero-pads columns n >= nvalid (for the 4-wide head output).
DEV v16h load_B_nk_pad(const _Float16* p, int ld, int n0, int k0, int nvalid) {
  int lane = threadIdx.x & 31;
  int n = n0 + (lane & 15);
  v16h b;
  if (n >= nvalid) {
#pragma unroll
    for (int i = 0; i < 16; ++i) b[i] = (_Float16)0.f;
    return b;
  }
  const _Float16* rp = p + n * ld + k0 + ((lane >> 4) << 4);
#pragma unroll
  for (int v = 0; v < 8; ++v) { b[2 * v] = rp[2 * v]; b[2 * v + 1] = rp[2 * v + 1]; }
  return b;
}

// B tile where the source matrix is row-major [K,N] (e.g. attention V).
DEV v16h load_B_kn(const _Float16* p, int ld, int n0, int k0) {
  int lane = threadIdx.x & 31;
  int n = n0 + (lane & 15);
  int kb = k0 + ((lane >> 4) << 4);
  v16h b;
#pragma unroll
  for (int v = 0; v < 8; ++v) {
    b[2 * v]     = p[(kb + 2 * v) * ld + n];
    b[2 * v + 1] = p[(kb + 2 * v + 1) * ld + n];
  }
  return b;
}

// ---------------------------------------------------------------------------
// Prep kernels
// ---------------------------------------------------------------------------
__global__ void cvt_f32_to_f16(const float* __restrict__ s, _Float16* __restrict__ d, int n) {
  int i = blockIdx.x * 256 + threadIdx.x;
  if (i < n) d[i] = (_Float16)s[i];
}

__global__ void build_x16(const float* __restrict__ z, const float* __restrict__ a,
                          _Float16* __restrict__ x16) {
  int i = blockIdx.x * 256 + threadIdx.x;      // over B*NA*DD
  int d = i & (DD - 1);
  int rn = i >> 7;                             // b*NA + na
  float v = (d < DZ) ? z[rn * DZ + d] : a[rn * DA + (d - DZ)];
  x16[i] = (_Float16)v;
}

__global__ void zero_acc(float* acc) {
  if (threadIdx.x < 17) acc[threadIdx.x] = 0.f;
}

// ---------------------------------------------------------------------------
// Router: logits = x_flat @ w_gate; top-2 softmax gates; balance-loss stats
// ---------------------------------------------------------------------------
__global__ __launch_bounds__(256) void router_kernel(
    const _Float16* __restrict__ x16, const float* __restrict__ wg,
    float* __restrict__ gates_ws, float* __restrict__ gates_out,
    float* __restrict__ acc) {
  __shared__ float buf[EE * 256];
  int tid = threadIdx.x, b = blockIdx.x;
  float p[EE];
#pragma unroll
  for (int e = 0; e < EE; ++e) p[e] = 0.f;
  const _Float16* xr = x16 + (size_t)b * TOKD;
  for (int k = tid; k < TOKD; k += 256) {
    float xv = (float)xr[k];
    const float* wr = wg + k * EE;
#pragma unroll
    for (int e = 0; e < EE; ++e) p[e] += xv * wr[e];
  }
#pragma unroll
  for (int e = 0; e < EE; ++e) buf[e * 256 + tid] = p[e];
  __syncthreads();
  for (int s = 128; s > 0; s >>= 1) {
    if (tid < s)
#pragma unroll
      for (int e = 0; e < EE; ++e) buf[e * 256 + tid] += buf[e * 256 + tid + s];
    __syncthreads();
  }
  if (tid == 0) {
    float lg[EE];
#pragma unroll
    for (int e = 0; e < EE; ++e) lg[e] = buf[e * 256];
    int i1 = 0;
    for (int e = 1; e < EE; ++e) if (lg[e] > lg[i1]) i1 = e;
    int i2 = -1;
    for (int e = 0; e < EE; ++e) if (e != i1 && (i2 < 0 || lg[e] > lg[i2])) i2 = e;
    float a2 = __expf(lg[i2] - lg[i1]);
    float s = 1.f + a2;
    float g1 = 1.f / s, g2 = a2 / s;
    float mx = lg[i1], se = 0.f;
#pragma unroll
    for (int e = 0; e < EE; ++e) se += __expf(lg[e] - mx);
    float lse = mx + __logf(se);
#pragma unroll
    for (int e = 0; e < EE; ++e) {
      float g = (e == i1) ? g1 : (e == i2) ? g2 : 0.f;
      gates_ws[b * EE + e] = g;
      gates_out[b * EE + e] = g;
      if (g > 0.f) { atomicAdd(&acc[e], g); atomicAdd(&acc[EE + e], 1.f); }
    }
    atomicAdd(&acc[16], lse);
  }
}

// ---------------------------------------------------------------------------
// Expert kernel: one block (8 waves) per token; loops over active experts.
// LDS map (dynamic, 131072 B total; XLS MUST be at offset 0 for the TDM D#):
//   [0      ) XLS   f16 32x128   8192
//   [8192   ) YACC  f32 32x128  16384
//   [24576  ) RES   f32 32x128  16384
//   [40960  ) SCB   f32 4x32x32 /32x128 16384
//   [57344  ) H16   f16 32x128   8192
//   [65536  ) union: {QKV f16 32x384 (24576) | FFN f16 32x1024 (65536)}
//             ATT f16 4x32x32 @ +24576 (8192), O16 f16 32x128 @ +32768 (8192)
// ---------------------------------------------------------------------------
__global__ __launch_bounds__(256) void expert_kernel(
    const _Float16* __restrict__ x16, const float* __restrict__ gates,
    const _Float16* __restrict__ w_in16, const float* __restrict__ b_in,
    const _Float16* __restrict__ w_out16, const float* __restrict__ b_out,
    const float* __restrict__ ln1g, const float* __restrict__ ln1b,
    const _Float16* __restrict__ w1_16, const float* __restrict__ b1,
    const _Float16* __restrict__ w2_16, const float* __restrict__ b2,
    const float* __restrict__ ln2g, const float* __restrict__ ln2b,
    _Float16* __restrict__ y16) {
  extern __shared__ char smem[];
  _Float16* XLS = (_Float16*)(smem);
  float*    YACC = (float*)(smem + 8192);
  float*    RES = (float*)(smem + 24576);
  float*    SCB = (float*)(smem + 40960);
  _Float16* H16 = (_Float16*)(smem + 57344);
  _Float16* QKV = (_Float16*)(smem + 65536);
  _Float16* ATT = (_Float16*)(smem + 65536 + 24576);
  _Float16* O16 = (_Float16*)(smem + 65536 + 32768);
  _Float16* FFN = (_Float16*)(smem + 65536);

  const int tid = threadIdx.x, wave = tid >> 5, lane = tid & 31;
  const int nn = lane & 15, mb = (lane >> 4) << 3;
  const int b = blockIdx.x;

#if defined(HAVE_TDM)
  // --- TDM: DMA the contiguous 8KB x-tile straight into LDS (wave 0) ------
  if (wave == 0) {
    unsigned long long ga = (unsigned long long)(const void*)(x16 + (size_t)b * TOKD);
    v4u g0;
    g0[0] = 1u;                                         // count=1, user mode
    g0[1] = 0u;                                         // lds_addr = 0 (XLS)
    g0[2] = (unsigned)(ga & 0xffffffffu);               // global_addr[31:0]
    g0[3] = (unsigned)((ga >> 32) & 0x01ffffffu)        // global_addr[56:32]
          | (2u << 30);                                 // type = 2 ("image")
    v8i g1;
#pragma unroll
    for (int i = 0; i < 8; ++i) g1[i] = 0;
    g1[0] = (3 << 16);                                  // data_size = 8 bytes
    g1[1] = (int)(1024u << 16);                         // tensor_dim0[15:0] @bit48
    g1[2] = (int)(1u << 16);                            // dim0 hi=0 | tensor_dim1=1
    g1[3] = (int)(1024u << 16);                         // dim1 hi=0 | tile_dim0=1024
    g1[4] = 1;                                          // tile_dim1=1, tile_dim2=0
    g1[5] = 1024;                                       // tensor_dim0_stride
    v4i gz;
#pragma unroll
    for (int i = 0; i < 4; ++i) gz[i] = 0;
    TDM_LOAD(g0, g1, gz, gz);
    __builtin_amdgcn_s_wait_tensorcnt(0);
  }
  for (int i = tid; i < NA * DD; i += 256) YACC[i] = 0.f;
#else
  for (int i = tid; i < NA * DD; i += 256) {
    XLS[i] = x16[(size_t)b * TOKD + i];
    YACC[i] = 0.f;
  }
#endif
  __syncthreads();

  for (int e = 0; e < EE; ++e) {
    float g = gates[b * EE + e];
    if (g == 0.f) continue;
    const _Float16* Wi = w_in16 + e * (3 * DD) * DD;
    const float*    Bi = b_in + e * (3 * DD);
    const _Float16* Wo = w_out16 + e * DD * DD;
    const float*    Bo = b_out + e * DD;
    const _Float16* W1 = w1_16 + e * FF * DD;
    const float*    B1 = b1 + e * FF;
    const _Float16* W2 = w2_16 + e * DD * FF;
    const float*    B2 = b2 + e * DD;
    const float* G1 = ln1g + e * DD; const float* Be1 = ln1b + e * DD;
    const float* G2 = ln2g + e * DD; const float* Be2 = ln2b + e * DD;

    // --- QKV = x @ Wi^T + Bi : [32,384], 48 tiles ---
    for (int t = wave; t < 48; t += 8) {
      int m = t & 1, n = t >> 1;
      v8f c = {};
#pragma unroll
      for (int ks = 0; ks < 4; ++ks) {
        v16h A = load_A(XLS, DD, m * 16, ks * 32);
        v16h Bv = load_B_nk(Wi, DD, n * 16, ks * 32);
        c = wmma16(A, Bv, c);
      }
#pragma unroll
      for (int r = 0; r < 8; ++r)
        QKV[(m * 16 + mb + r) * 384 + n * 16 + nn] = (_Float16)(c[r] + Bi[n * 16 + nn]);
    }
    __syncthreads();

    // --- scores = Q @ K^T per head : 4 heads x (2x2 tiles) ---
    for (int t = wave; t < 16; t += 8) {
      int h = t >> 2, m = (t >> 1) & 1, n = t & 1;
      v8f c = {};
      v16h A = load_A(QKV, 384, m * 16, h * HD);
      v16h Bv = load_B_nk(QKV + DD + h * HD, 384, n * 16, 0);
      c = wmma16(A, Bv, c);
#pragma unroll
      for (int r = 0; r < 8; ++r)
        SCB[h * 1024 + (m * 16 + mb + r) * 32 + n * 16 + nn] = c[r];
    }
    __syncthreads();

    // --- softmax (scaled by 1/sqrt(32)), 128 rows ---
    if (tid < NH * NA) {
      int h = tid >> 5, row = tid & 31;
      float* sr = SCB + h * 1024 + row * 32;
      const float scale = 0.17677669529663687f;
      float mx = sr[0];
#pragma unroll
      for (int k = 1; k < 32; ++k) mx = fmaxf(mx, sr[k]);
      float s = 0.f;
#pragma unroll
      for (int k = 0; k < 32; ++k) { float ev = __expf((sr[k] - mx) * scale); sr[k] = ev; s += ev; }
      float inv = 1.f / s;
#pragma unroll
      for (int k = 0; k < 32; ++k)
        ATT[h * 1024 + row * 32 + k] = (_Float16)(sr[k] * inv);
    }
    __syncthreads();

    // --- O = att @ V per head ---
    for (int t = wave; t < 16; t += 8) {
      int h = t >> 2, m = (t >> 1) & 1, n = t & 1;
      v8f c = {};
      v16h A = load_A(ATT + h * 1024, 32, m * 16, 0);
      v16h Bv = load_B_kn(QKV + 2 * DD + h * HD, 384, n * 16, 0);
      c = wmma16(A, Bv, c);
#pragma unroll
      for (int r = 0; r < 8; ++r)
        O16[(m * 16 + mb + r) * DD + h * HD + n * 16 + nn] = (_Float16)c[r];
    }
    __syncthreads();

    // --- RES = x + O @ Wo^T + Bo ---
    for (int t = wave; t < 16; t += 8) {
      int m = t & 1, n = t >> 1;
      v8f c = {};
#pragma unroll
      for (int ks = 0; ks < 4; ++ks) {
        v16h A = load_A(O16, DD, m * 16, ks * 32);
        v16h Bv = load_B_nk(Wo, DD, n * 16, ks * 32);
        c = wmma16(A, Bv, c);
      }
#pragma unroll
      for (int r = 0; r < 8; ++r) {
        int row = m * 16 + mb + r, col = n * 16 + nn;
        RES[row * DD + col] = (float)XLS[row * DD + col] + c[r] + Bo[col];
      }
    }
    __syncthreads();

    // --- LN1 -> RES(f32) + H16(f16) ---
    if (tid < NA) {
      float* rr = RES + tid * DD;
      float mu = 0.f;
      for (int c = 0; c < DD; ++c) mu += rr[c];
      mu *= (1.f / DD);
      float var = 0.f;
      for (int c = 0; c < DD; ++c) { float d = rr[c] - mu; var += d * d; }
      var *= (1.f / DD);
      float rinv = rsqrtf(var + 1e-5f);
      for (int c = 0; c < DD; ++c) {
        float v = (rr[c] - mu) * rinv * G1[c] + Be1[c];
        rr[c] = v;
        H16[tid * DD + c] = (_Float16)v;
      }
    }
    __syncthreads();

    // --- FFN1: relu(h @ W1^T + B1) : [32,1024], 128 tiles ---
    for (int t = wave; t < 128; t += 8) {
      int m = t & 1, n = t >> 1;
      v8f c = {};
#pragma unroll
      for (int ks = 0; ks < 4; ++ks) {
        v16h A = load_A(H16, DD, m * 16, ks * 32);
        v16h Bv = load_B_nk(W1, DD, n * 16, ks * 32);
        c = wmma16(A, Bv, c);
      }
#pragma unroll
      for (int r = 0; r < 8; ++r) {
        float v = c[r] + B1[n * 16 + nn];
        FFN[(m * 16 + mb + r) * FF + n * 16 + nn] = (_Float16)(v > 0.f ? v : 0.f);
      }
    }
    __syncthreads();

    // --- FFN2: f @ W2^T + B2 -> SCB(f32 [32,128]) ---
    for (int t = wave; t < 16; t += 8) {
      int m = t & 1, n = t >> 1;
      v8f c = {};
      for (int ks = 0; ks < 32; ++ks) {
        v16h A = load_A(FFN, FF, m * 16, ks * 32);
        v16h Bv = load_B_nk(W2, FF, n * 16, ks * 32);
        c = wmma16(A, Bv, c);
      }
#pragma unroll
      for (int r = 0; r < 8; ++r)
        SCB[(m * 16 + mb + r) * DD + n * 16 + nn] = c[r] + B2[n * 16 + nn];
    }
    __syncthreads();

    // --- LN2(h + f), gate-weighted accumulate into YACC (no private array) ---
    if (tid < NA) {
      const float* r1 = RES + tid * DD;
      const float* r2 = SCB + tid * DD;
      float mu = 0.f;
      for (int c = 0; c < DD; ++c) mu += r1[c] + r2[c];
      mu *= (1.f / DD);
      float var = 0.f;
      for (int c = 0; c < DD; ++c) { float d = r1[c] + r2[c] - mu; var += d * d; }
      var *= (1.f / DD);
      float rinv = rsqrtf(var + 1e-5f);
      for (int c = 0; c < DD; ++c)
        YACC[tid * DD + c] += g * ((r1[c] + r2[c] - mu) * rinv * G2[c] + Be2[c]);
    }
    __syncthreads();
  }

  for (int i = tid; i < NA * DD; i += 256)
    y16[(size_t)b * TOKD + i] = (_Float16)YACC[i];
}

// ---------------------------------------------------------------------------
// Head: r = relu(y_flat @ hw1^T + hb1) @ hw2^T + hb2   (32 tokens / block)
// ---------------------------------------------------------------------------
__global__ __launch_bounds__(256) void head_kernel(
    const _Float16* __restrict__ y16, const _Float16* __restrict__ hw1,
    const float* __restrict__ hb1, const _Float16* __restrict__ hw2,
    const float* __restrict__ hb2, float* __restrict__ r_out) {
  __shared__ _Float16 HHs[32 * HH];
  const int tid = threadIdx.x, wave = tid >> 5, lane = tid & 31;
  const int nn = lane & 15, mb = (lane >> 4) << 3;
  const int b0 = blockIdx.x * 32;
  const int m = wave & 1, ng = wave >> 1;   // ng in 0..3 -> 8 N-tiles each

  v8f acc[8];
#pragma unroll
  for (int j = 0; j < 8; ++j) { v8f zv = {}; acc[j] = zv; }

  for (int ks = 0; ks < TOKD / 32; ++ks) {
    v16h A = load_A(y16 + (size_t)b0 * TOKD, TOKD, m * 16, ks * 32);
#pragma unroll
    for (int j = 0; j < 8; ++j) {
      int nt = ng * 8 + j;
      v16h Bv = load_B_nk(hw1, TOKD, nt * 16, ks * 32);
      acc[j] = wmma16(A, Bv, acc[j]);
    }
  }
#pragma unroll
  for (int j = 0; j < 8; ++j) {
    int nt = ng * 8 + j;
#pragma unroll
    for (int r = 0; r < 8; ++r) {
      float v = acc[j][r] + hb1[nt * 16 + nn];
      HHs[(m * 16 + mb + r) * HH + nt * 16 + nn] = (_Float16)(v > 0.f ? v : 0.f);
    }
  }
  __syncthreads();

  if (wave < 2) {
    int mm = wave;
    v8f c = {};
    for (int ks = 0; ks < HH / 32; ++ks) {
      v16h A = load_A(HHs, HH, mm * 16, ks * 32);
      v16h Bv = load_B_nk_pad(hw2, HH, 0, ks * 32, NRR);
      c = wmma16(A, Bv, c);
    }
    if (nn < NRR) {
#pragma unroll
      for (int r = 0; r < 8; ++r)
        r_out[(b0 + mm * 16 + mb + r) * NRR + nn] = c[r] + hb2[nn];
    }
  }
}

// ---------------------------------------------------------------------------
// Finalize balance loss: cv2(importance) + cv2(load) + mean(logsumexp)
// ---------------------------------------------------------------------------
__global__ void finalize_kernel(const float* __restrict__ acc, float* __restrict__ out_loss) {
  if (threadIdx.x == 0 && blockIdx.x == 0) {
    float mi = 0.f, ml = 0.f;
    for (int e = 0; e < EE; ++e) { mi += acc[e]; ml += acc[EE + e]; }
    mi *= (1.f / EE); ml *= (1.f / EE);
    float vi = 0.f, vl = 0.f;
    for (int e = 0; e < EE; ++e) {
      float d = acc[e] - mi; vi += d * d;
      d = acc[EE + e] - ml; vl += d * d;
    }
    vi *= (1.f / (EE - 1)); vl *= (1.f / (EE - 1));
    *out_loss = vi / (mi * mi + 1e-10f) + vl / (ml * ml + 1e-10f) + acc[16] / (float)BB;
  }
}

// ---------------------------------------------------------------------------
// Launch
// ---------------------------------------------------------------------------
extern "C" void kernel_launch(void* const* d_in, const int* in_sizes, int n_in,
                              void* d_out, int out_size, void* d_ws, size_t ws_size,
                              hipStream_t stream) {
  const float* z    = (const float*)d_in[0];
  const float* a    = (const float*)d_in[1];
  const float* wg   = (const float*)d_in[2];
  const float* w_in = (const float*)d_in[3];
  const float* b_in = (const float*)d_in[4];
  const float* w_out= (const float*)d_in[5];
  const float* b_out= (const float*)d_in[6];
  const float* ln1g = (const float*)d_in[7];
  const float* ln1b = (const float*)d_in[8];
  const float* w1   = (const float*)d_in[9];
  const float* b1   = (const float*)d_in[10];
  const float* w2   = (const float*)d_in[11];
  const float* b2   = (const float*)d_in[12];
  const float* ln2g = (const float*)d_in[13];
  const float* ln2b = (const float*)d_in[14];
  const float* hw1  = (const float*)d_in[15];
  const float* hb1  = (const float*)d_in[16];
  const float* hw2  = (const float*)d_in[17];
  const float* hb2  = (const float*)d_in[18];

  // workspace layout (bytes)
  constexpr size_t N_X   = (size_t)BB * TOKD;          // 4,194,304
  constexpr size_t N_WIN = (size_t)EE * 3 * DD * DD;   // 393,216
  constexpr size_t N_WOUT= (size_t)EE * DD * DD;       // 131,072
  constexpr size_t N_W1  = (size_t)EE * FF * DD;       // 1,048,576
  constexpr size_t N_W2  = (size_t)EE * DD * FF;       // 1,048,576
  constexpr size_t N_HW1 = (size_t)HH * TOKD;          // 2,097,152
  constexpr size_t N_HW2 = (size_t)NRR * HH;           // 2,048

  char* ws = (char*)d_ws;
  size_t off = 0;
  _Float16* x16    = (_Float16*)(ws + off); off += N_X * 2;
  _Float16* y16    = (_Float16*)(ws + off); off += N_X * 2;
  _Float16* win16  = (_Float16*)(ws + off); off += N_WIN * 2;
  _Float16* wout16 = (_Float16*)(ws + off); off += N_WOUT * 2;
  _Float16* w1_16  = (_Float16*)(ws + off); off += N_W1 * 2;
  _Float16* w2_16  = (_Float16*)(ws + off); off += N_W2 * 2;
  _Float16* hw1_16 = (_Float16*)(ws + off); off += N_HW1 * 2;
  _Float16* hw2_16 = (_Float16*)(ws + off); off += N_HW2 * 2;
  float*    gates_ws = (float*)(ws + off); off += (size_t)BB * EE * 4;
  float*    accp     = (float*)(ws + off); off += 17 * 4;

  float* r_out    = (float*)d_out;                  // [1024,4]
  float* g_out    = (float*)d_out + BB * NRR;       // [1024,8]
  float* loss_out = (float*)d_out + BB * NRR + BB * EE;

  // prep
  build_x16<<<(N_X + 255) / 256, 256, 0, stream>>>(z, a, x16);
  cvt_f32_to_f16<<<(N_WIN + 255) / 256, 256, 0, stream>>>(w_in, win16, (int)N_WIN);
  cvt_f32_to_f16<<<(N_WOUT + 255) / 256, 256, 0, stream>>>(w_out, wout16, (int)N_WOUT);
  cvt_f32_to_f16<<<(N_W1 + 255) / 256, 256, 0, stream>>>(w1, w1_16, (int)N_W1);
  cvt_f32_to_f16<<<(N_W2 + 255) / 256, 256, 0, stream>>>(w2, w2_16, (int)N_W2);
  cvt_f32_to_f16<<<(N_HW1 + 255) / 256, 256, 0, stream>>>(hw1, hw1_16, (int)N_HW1);
  cvt_f32_to_f16<<<(N_HW2 + 255) / 256, 256, 0, stream>>>(hw2, hw2_16, (int)N_HW2);
  zero_acc<<<1, 32, 0, stream>>>(accp);

  // router -> gates + stats
  router_kernel<<<BB, 256, 0, stream>>>(x16, wg, gates_ws, g_out, accp);

  // experts (1 block per token, 128 KB dynamic LDS)
  expert_kernel<<<BB, 256, 131072, stream>>>(x16, gates_ws,
      win16, b_in, wout16, b_out, ln1g, ln1b,
      w1_16, b1, w2_16, b2, ln2g, ln2b, y16);

  // reward head
  head_kernel<<<BB / 32, 256, 0, stream>>>(y16, hw1_16, hb1, hw2_16, hb2, r_out);

  // balance loss scalar
  finalize_kernel<<<1, 64, 0, stream>>>(accp, loss_out);
}
